// adaifl_model_61306363183251
// MI455X (gfx1250) — matrix operand order repaired and checked by reading, not compiled, not run on hardware
//
#include <hip/hip_runtime.h>
#include <hip/hip_bf16.h>

// ---------------- CDNA5 WMMA types ----------------
typedef __attribute__((ext_vector_type(16))) __bf16 v16bf;
typedef __attribute__((ext_vector_type(8)))  float  v8f;

__device__ inline unsigned short f2bf(float f) {
    unsigned u = __float_as_uint(f);
    unsigned r = u + 0x7fffu + ((u >> 16) & 1u);   // round-to-nearest-even
    return (unsigned short)(r >> 16);
}

__device__ inline v8f wmma_bf16(v16bf a, v16bf b, v8f c) {
    return __builtin_amdgcn_wmma_f32_16x16x32_bf16(false, a, false, b, (short)0, c, false, false);
}

// A-matrix fragment 16x32 bf16 from row-major [row][k] buffer (ld in elements).
// Lane m = lane&15, half = lane>>4. Elements 0..7 -> K = k0+8*half+j,
// elements 8..15 -> K = k0+16+8*half+j  (per ISA 16-bit A layout).
__device__ inline v16bf load_frag_a(const unsigned short* p, int ld, int row0, int k0) {
    int lane = threadIdx.x & 31;
    int m = lane & 15, half = lane >> 4;
    const unsigned short* r = p + (size_t)(row0 + m) * ld + k0 + 8 * half;
    union { v16bf v; uint4 q[2]; } u;
    u.q[0] = *(const uint4*)(r);
    u.q[1] = *(const uint4*)(r + 16);
    return u.v;
}

// B-matrix fragment 32x16 bf16 from [N][K] row-major buffer (B transposed).
// Lane n = lane&15, half = lane>>4. Elements 0..7 -> K = k0+8*half+j,
// elements 8..15 -> K = k0+16+8*half+j (same as A).
__device__ inline v16bf load_frag_b(const unsigned short* p, int ld, int n0, int k0) {
    int lane = threadIdx.x & 31;
    int n = lane & 15, half = lane >> 4;
    const unsigned short* r = p + (size_t)(n0 + n) * ld + k0 + 8 * half;
    union { v16bf v; uint4 q[2]; } u;
    u.q[0] = *(const uint4*)(r);
    u.q[1] = *(const uint4*)(r + 16);
    return u.v;
}

__device__ inline float rmax16(float v) {
    for (int o = 8; o >= 1; o >>= 1) v = fmaxf(v, __shfl_xor(v, o, 16));
    return v;
}
__device__ inline float rsum16(float v) {
    for (int o = 8; o >= 1; o >>= 1) v += __shfl_xor(v, o, 16);
    return v;
}

// ---------------- gfx1250 async copy (global -> LDS), tracked by ASYNCcnt -----
// Stage a 128x32 bf16 tile (rows tile0..tile0+127, cols k0..k0+31 of a row-major
// [*, ldK] buffer) into LDS. 256 threads; each issues two 16B async copies.
__device__ inline void async_tile_load_128x32(const unsigned short* gbase, int ldK,
                                              int tile0, int k0,
                                              unsigned short* lds) {
#pragma unroll
    for (int p = 0; p < 2; ++p) {
        int c = (int)threadIdx.x + p * 256;          // 0..511 chunk id
        int row = c >> 2, seg = c & 3;               // 4 x 8-element segs per row
        unsigned ga = ((unsigned)(tile0 + row) * (unsigned)ldK
                       + (unsigned)(k0 + seg * 8)) * 2u;          // byte offset
        unsigned la = (unsigned)(uintptr_t)(lds + row * 32 + seg * 8); // LDS offset
        asm volatile("global_load_async_to_lds_b128 %0, %1, %2"
                     :: "v"(la), "v"(ga), "s"(gbase) : "memory");
    }
}
__device__ inline void wait_async_le4() {
    asm volatile("s_wait_asynccnt 0x4" ::: "memory");
}
__device__ inline void wait_async_0() {
    asm volatile("s_wait_asynccnt 0x0" ::: "memory");
}

// ---------------- constants ----------------
#define HW   2304
#define CDIM 768
#define NH   12
#define HD   64
#define W48  48
#define MLPD 3072

// ---------------- layernorm -> bf16 ----------------
__global__ void k_ln(const float* __restrict__ x, const float* __restrict__ g,
                     const float* __restrict__ b, unsigned short* __restrict__ out) {
    const int C = CDIM;
    int row = blockIdx.x;
    const float* xr = x + (size_t)row * C;
    __shared__ float red[256];
    int t = threadIdx.x;
    float s = 0.f;
    for (int c = t; c < C; c += 256) s += xr[c];
    red[t] = s; __syncthreads();
    for (int o = 128; o > 0; o >>= 1) { if (t < o) red[t] += red[t + o]; __syncthreads(); }
    float mu = red[0] / C; __syncthreads();
    float v = 0.f;
    for (int c = t; c < C; c += 256) { float d = xr[c] - mu; v += d * d; }
    red[t] = v; __syncthreads();
    for (int o = 128; o > 0; o >>= 1) { if (t < o) red[t] += red[t + o]; __syncthreads(); }
    float inv = rsqrtf(red[0] / C + 1e-5f);
    for (int c = t; c < C; c += 256)
        out[(size_t)row * C + c] = f2bf((xr[c] - mu) * inv * g[c] + b[c]);
}

// ---------------- fp32 [K][N] -> bf16 [N][K] (transpose + convert) ----------------
__global__ void k_cvt_t(const float* __restrict__ W, unsigned short* __restrict__ Wt,
                        int K, int N) {
    long idx = (long)blockIdx.x * blockDim.x + threadIdx.x;
    if (idx >= (long)K * N) return;
    int n = (int)(idx / K), k = (int)(idx % K);
    Wt[idx] = f2bf(W[(size_t)k * N + n]);
}

// ---------------- WMMA GEMM: out[M][N] = A_bf16[M][K] * Bt_bf16[N][K] ----------------
// Block = 256 threads (8 waves) -> 128x128 output tile. K staged in 32-wide
// double-buffered LDS tiles filled by gfx1250 async copies; prefetch of tile
// k+1 overlaps WMMA compute on tile k. Each wave: 16 rows x 128 cols.
// Epilogue: +bias, optional exact GELU, optional +residual, fp32/bf16 stores.
__global__ void __launch_bounds__(256)
k_gemm(const unsigned short* __restrict__ A,
       const unsigned short* __restrict__ Bt,
       const float* __restrict__ bias,
       const float* __restrict__ residual,
       float* __restrict__ outf, unsigned short* __restrict__ outb,
       int M, int N, int K, int act) {
    __shared__ __align__(16) unsigned short Abuf[2][128 * 32];
    __shared__ __align__(16) unsigned short Bbuf[2][128 * 32];

    int tileM = blockIdx.x * 128;
    int nbase = blockIdx.y * 128;
    int wave = threadIdx.x >> 5;
    int rw0 = wave * 16;                       // wave's 16 rows within the tile

    v8f acc[8];
    const v8f zero8 = {0.f, 0.f, 0.f, 0.f, 0.f, 0.f, 0.f, 0.f};
#pragma unroll
    for (int f = 0; f < 8; ++f) acc[f] = zero8;

    int nk = K >> 5;
    async_tile_load_128x32(A,  K, tileM, 0, &Abuf[0][0]);
    async_tile_load_128x32(Bt, K, nbase, 0, &Bbuf[0][0]);

    for (int kt = 0; kt < nk; ++kt) {
        int buf = kt & 1;
        if (kt + 1 < nk) {
            async_tile_load_128x32(A,  K, tileM, (kt + 1) * 32, &Abuf[buf ^ 1][0]);
            async_tile_load_128x32(Bt, K, nbase, (kt + 1) * 32, &Bbuf[buf ^ 1][0]);
            wait_async_le4();                  // in-order: current tile's 4 done
        } else {
            wait_async_0();
        }
        __syncthreads();                       // tile[buf] visible to all waves

        const unsigned short* Ab = &Abuf[buf][0];
        const unsigned short* Bb = &Bbuf[buf][0];
        v16bf a = load_frag_a(Ab, 32, rw0, 0);
#pragma unroll
        for (int f = 0; f < 8; ++f)
            acc[f] = wmma_bf16(a, load_frag_b(Bb, 32, f * 16, 0), acc[f]);

        __syncthreads();                       // done reading buf before reuse
    }

    int lane = threadIdx.x & 31;
    int n = lane & 15, mh = lane >> 4;
#pragma unroll
    for (int f = 0; f < 8; ++f) {
        int col = nbase + f * 16 + n;
        float bv = bias ? bias[col] : 0.f;
#pragma unroll
        for (int r = 0; r < 8; ++r) {
            int m = tileM + rw0 + r + 8 * mh;
            float v = acc[f][r] + bv;
            if (act == 1) v = 0.5f * v * (1.f + erff(v * 0.70710678118654752f));
            if (residual) v += residual[(size_t)m * N + col];
            if (outf) outf[(size_t)m * N + col] = v;
            if (outb) outb[(size_t)m * N + col] = f2bf(v);
        }
    }
}

// ---------------- split qkv fp32 [HW][3C] -> q_bf16 (scaled), k_bf16, vT_bf16 ----------------
__global__ void k_repack(const float* __restrict__ qkv,
                         unsigned short* __restrict__ qb,
                         unsigned short* __restrict__ kb,
                         unsigned short* __restrict__ vT) {
    int idx = blockIdx.x * blockDim.x + threadIdx.x;
    if (idx >= HW * CDIM) return;
    int q = idx / CDIM, c = idx % CDIM;
    int h = c >> 6, d = c & 63;
    size_t rowb = (size_t)q * (3 * CDIM);
    float qv = qkv[rowb + c];
    float kv = qkv[rowb + CDIM + c];
    float vv = qkv[rowb + 2 * CDIM + c];
    qb[((size_t)h * HW + q) * HD + d] = f2bf(qv * 0.125f);   // scale = hd^-0.5 = 1/8
    kb[((size_t)h * HW + q) * HD + d] = f2bf(kv);
    vT[((size_t)h * HD + d) * HW + q] = f2bf(vv);
}

// ---------------- decomposed rel-pos bias (uses UNscaled q, fp32) ----------------
__global__ void k_relbias(const float* __restrict__ qkv,
                          const float* __restrict__ rph, const float* __restrict__ rpw,
                          float* __restrict__ relh, float* __restrict__ relw) {
    int idx = blockIdx.x * blockDim.x + threadIdx.x;
    if (idx >= NH * HW * W48) return;
    int k = idx % W48;
    int t = idx / W48;
    int q = t % HW;
    int h = t / HW;
    int qh = q / W48, qw = q % W48;
    const float* qv = qkv + (size_t)q * (3 * CDIM) + h * HD;
    const float* Rh = rph + (size_t)(qh - k + W48 - 1) * HD;
    const float* Rw = rpw + (size_t)(qw - k + W48 - 1) * HD;
    float sh = 0.f, sw = 0.f;
    for (int d = 0; d < HD; ++d) { float xv = qv[d]; sh += xv * Rh[d]; sw += xv * Rw[d]; }
    relh[idx] = sh;
    relw[idx] = sw;
}

// ---------------- fused flash-attention (one wave per head x 16-query tile) ----------------
__global__ void k_attn(const unsigned short* __restrict__ qb,
                       const unsigned short* __restrict__ kb,
                       const unsigned short* __restrict__ vT,
                       const float* __restrict__ relh, const float* __restrict__ relw,
                       unsigned short* __restrict__ outb) {
    int head = blockIdx.y;
    int row0 = blockIdx.x * 16;
    const unsigned short* qh = qb + (size_t)head * HW * HD;
    const unsigned short* kh = kb + (size_t)head * HW * HD;
    const unsigned short* vh = vT + (size_t)head * HD * HW;
    const float* rh = relh + (size_t)head * HW * W48;
    const float* rw = relw + (size_t)head * HW * W48;
    __shared__ __align__(16) unsigned short P[16 * 32];

    int lane = threadIdx.x & 31;
    int n = lane & 15, mh = lane >> 4;

    v16bf aq0 = load_frag_a(qh, HD, row0, 0);    // Q loaded once, reused all key tiles
    v16bf aq1 = load_frag_a(qh, HD, row0, 32);

    const v8f zero8 = {0.f, 0.f, 0.f, 0.f, 0.f, 0.f, 0.f, 0.f};
    v8f o[4]; o[0] = zero8; o[1] = zero8; o[2] = zero8; o[3] = zero8;
    float mrow[8], lrow[8];
    for (int r = 0; r < 8; ++r) { mrow[r] = -3.0e38f; lrow[r] = 0.f; }

    for (int key0 = 0; key0 < HW; key0 += 32) {
        v8f s0 = zero8, s1 = zero8;
        s0 = wmma_bf16(aq0, load_frag_b(kh, HD, key0,      0),  s0);
        s0 = wmma_bf16(aq1, load_frag_b(kh, HD, key0,      32), s0);
        s1 = wmma_bf16(aq0, load_frag_b(kh, HD, key0 + 16, 0),  s1);
        s1 = wmma_bf16(aq1, load_frag_b(kh, HD, key0 + 16, 32), s1);

        int keyA = key0 + n, keyB = key0 + 16 + n;
        int khA = keyA / W48, kwA = keyA % W48;
        int khB = keyB / W48, kwB = keyB % W48;

        for (int r = 0; r < 8; ++r) {
            int q = row0 + r + 8 * mh;
            float v0 = s0[r] + rh[(size_t)q * W48 + khA] + rw[(size_t)q * W48 + kwA];
            float v1 = s1[r] + rh[(size_t)q * W48 + khB] + rw[(size_t)q * W48 + kwB];
            float mx = rmax16(fmaxf(v0, v1));
            float mnew = fmaxf(mrow[r], mx);
            float fac = __expf(mrow[r] - mnew);
            float p0 = __expf(v0 - mnew);
            float p1 = __expf(v1 - mnew);
            float psum = rsum16(p0 + p1);
            lrow[r] = lrow[r] * fac + psum;
            mrow[r] = mnew;
            o[0][r] *= fac; o[1][r] *= fac; o[2][r] *= fac; o[3][r] *= fac;
            int mt = r + 8 * mh;
            P[mt * 32 + n]      = f2bf(p0);
            P[mt * 32 + 16 + n] = f2bf(p1);
        }
        __syncthreads();
        v16bf ap = load_frag_a(P, 32, 0, 0);
        o[0] = wmma_bf16(ap, load_frag_b(vh, HW, 0,  key0), o[0]);
        o[1] = wmma_bf16(ap, load_frag_b(vh, HW, 16, key0), o[1]);
        o[2] = wmma_bf16(ap, load_frag_b(vh, HW, 32, key0), o[2]);
        o[3] = wmma_bf16(ap, load_frag_b(vh, HW, 48, key0), o[3]);
        __syncthreads();
    }

    for (int r = 0; r < 8; ++r) {
        int q = row0 + r + 8 * mh;
        float inv = 1.f / lrow[r];
        size_t base = (size_t)q * CDIM + head * HD;
        outb[base + 0  + n] = f2bf(o[0][r] * inv);
        outb[base + 16 + n] = f2bf(o[1][r] * inv);
        outb[base + 32 + n] = f2bf(o[2][r] * inv);
        outb[base + 48 + n] = f2bf(o[3][r] * inv);
    }
}

// ---------------- host launch ----------------
extern "C" void kernel_launch(void* const* d_in, const int* in_sizes, int n_in,
                              void* d_out, int out_size, void* d_ws, size_t ws_size,
                              hipStream_t stream) {
    const float* x      = (const float*)d_in[0];
    const float* qkv_w  = (const float*)d_in[1];
    const float* qkv_b  = (const float*)d_in[2];
    const float* proj_w = (const float*)d_in[3];
    const float* proj_b = (const float*)d_in[4];
    const float* rph    = (const float*)d_in[5];
    const float* rpw    = (const float*)d_in[6];
    const float* ln1_g  = (const float*)d_in[7];
    const float* ln1_b  = (const float*)d_in[8];
    const float* ln2_g  = (const float*)d_in[9];
    const float* ln2_b  = (const float*)d_in[10];
    const float* mlp_w1 = (const float*)d_in[11];
    const float* mlp_b1 = (const float*)d_in[12];
    const float* mlp_w2 = (const float*)d_in[13];
    const float* mlp_b2 = (const float*)d_in[14];
    float* out = (float*)d_out;

    char* wp = (char*)d_ws;
    auto alloc = [&](size_t bytes) -> void* {
        void* p = (void*)wp;
        wp += (bytes + 255) & ~(size_t)255;
        return p;
    };
    unsigned short* wqkvT = (unsigned short*)alloc((size_t)3 * CDIM * CDIM * 2);
    unsigned short* wprojT= (unsigned short*)alloc((size_t)CDIM * CDIM * 2);
    unsigned short* w1T   = (unsigned short*)alloc((size_t)MLPD * CDIM * 2);
    unsigned short* w2T   = (unsigned short*)alloc((size_t)CDIM * MLPD * 2);
    unsigned short* xnb   = (unsigned short*)alloc((size_t)HW * CDIM * 2);
    float*          qkvf  = (float*)        alloc((size_t)HW * 3 * CDIM * 4);
    unsigned short* qb    = (unsigned short*)alloc((size_t)NH * HW * HD * 2);
    unsigned short* kbuf  = (unsigned short*)alloc((size_t)NH * HW * HD * 2);
    unsigned short* vTb   = (unsigned short*)alloc((size_t)NH * HD * HW * 2);
    float*          relh  = (float*)        alloc((size_t)NH * HW * W48 * 4);
    float*          relw  = (float*)        alloc((size_t)NH * HW * W48 * 4);
    unsigned short* aob   = (unsigned short*)alloc((size_t)HW * CDIM * 2);
    float*          x2    = (float*)        alloc((size_t)HW * CDIM * 4);
    unsigned short* xn2b  = (unsigned short*)alloc((size_t)HW * CDIM * 2);
    unsigned short* hb    = (unsigned short*)alloc((size_t)HW * MLPD * 2);

    auto cdiv = [](long a, long b) { return (int)((a + b - 1) / b); };

    // weight transpose+convert
    k_cvt_t<<<cdiv((long)CDIM * 3 * CDIM, 256), 256, 0, stream>>>(qkv_w,  wqkvT, CDIM, 3 * CDIM);
    k_cvt_t<<<cdiv((long)CDIM * CDIM,     256), 256, 0, stream>>>(proj_w, wprojT, CDIM, CDIM);
    k_cvt_t<<<cdiv((long)CDIM * MLPD,     256), 256, 0, stream>>>(mlp_w1, w1T,    CDIM, MLPD);
    k_cvt_t<<<cdiv((long)MLPD * CDIM,     256), 256, 0, stream>>>(mlp_w2, w2T,    MLPD, CDIM);

    // LN1 -> bf16
    k_ln<<<HW, 256, 0, stream>>>(x, ln1_g, ln1_b, xnb);

    // QKV GEMM: [2304 x 768] * [768 x 2304]
    k_gemm<<<dim3(HW / 128, (3 * CDIM) / 128), 256, 0, stream>>>(
        xnb, wqkvT, qkv_b, nullptr, qkvf, nullptr, HW, 3 * CDIM, CDIM, 0);

    // split heads + scale q + transpose v
    k_repack<<<cdiv((long)HW * CDIM, 256), 256, 0, stream>>>(qkvf, qb, kbuf, vTb);

    // rel-pos bias tables
    k_relbias<<<cdiv((long)NH * HW * W48, 256), 256, 0, stream>>>(qkvf, rph, rpw, relh, relw);

    // fused flash attention -> bf16 [2304][768]
    k_attn<<<dim3(HW / 16, NH), 32, 0, stream>>>(qb, kbuf, vTb, relh, relw, aob);

    // proj GEMM + residual(x) -> x2 fp32
    k_gemm<<<dim3(HW / 128, CDIM / 128), 256, 0, stream>>>(
        aob, wprojT, proj_b, x, x2, nullptr, HW, CDIM, CDIM, 0);

    // LN2 -> bf16
    k_ln<<<HW, 256, 0, stream>>>(x2, ln2_g, ln2_b, xn2b);

    // MLP1 GEMM + exact GELU -> bf16 hidden
    k_gemm<<<dim3(HW / 128, MLPD / 128), 256, 0, stream>>>(
        xn2b, w1T, mlp_b1, nullptr, nullptr, hb, HW, MLPD, CDIM, 1);

    // MLP2 GEMM + residual(x2) -> final output
    k_gemm<<<dim3(HW / 128, CDIM / 128), 256, 0, stream>>>(
        hb, w2T, mlp_b2, x2, out, nullptr, HW, CDIM, MLPD, 0);
}
